// SoftDecisionTree_58325655879926
// MI455X (gfx1250) — matrix operand order repaired and checked by reading, not compiled, not run on hardware
//
#include <hip/hip_runtime.h>
#include <hip/hip_bf16.h>

typedef __attribute__((ext_vector_type(16))) __bf16 bf16x16;
typedef __attribute__((ext_vector_type(8)))  float  v8f;
typedef __attribute__((ext_vector_type(4)))  int    v4i;

// ---------------- constants for this problem ----------------
#define BATCH   8192
#define FEAT    4096
#define NCLS    1000
#define NCLSP   1024      // padded class dim for GEMM2
#define NODES   512       // 511 internal nodes padded to 512
#define NBIAS   511
#define LEAVES  512
#define LEVELS  9

// ---------------- CDNA5 async global->LDS path (guarded) ----------------
#if __has_builtin(__builtin_amdgcn_global_load_async_to_lds_b128)
#define HAVE_ASYNC_LDS 1
#else
#define HAVE_ASYNC_LDS 0
#endif

#define GLOBAL_AS __attribute__((address_space(1)))
#define LOCAL_AS  __attribute__((address_space(3)))

#if HAVE_ASYNC_LDS
#if __has_builtin(__builtin_amdgcn_s_wait_asynccnt)
#define WAIT_ASYNC() __builtin_amdgcn_s_wait_asynccnt(0)
#else
#define WAIT_ASYNC() asm volatile("s_wait_asynccnt 0x0" ::: "memory")
#endif
#endif

__device__ __forceinline__ unsigned short f2bf(float f) {
    unsigned u = __float_as_uint(f);
    u += 0x7FFFu + ((u >> 16) & 1u);          // round-to-nearest-even
    return (unsigned short)(u >> 16);
}

union FragBF { bf16x16 v; int4 q[2]; };

// ---------------- conversion kernels ----------------
__global__ void cvt_f32_bf16(const float* __restrict__ src,
                             unsigned short* __restrict__ dst, long n) {
    long i = (long)blockIdx.x * blockDim.x + threadIdx.x;
    if (i < n) dst[i] = f2bf(src[i]);
}

// W (511 x 4096) -> Wb (512 x 4096), row 511 zeroed
__global__ void cvt_W(const float* __restrict__ W,
                      unsigned short* __restrict__ Wb) {
    long i = (long)blockIdx.x * blockDim.x + threadIdx.x;   // over 512*4096
    long row = i >> 12;
    float v = (row < NBIAS) ? W[i] : 0.0f;
    Wb[i] = f2bf(v);
}

// softmax over leaf_params rows, stored TRANSPOSED as (NCLSP x LEAVES) bf16
__global__ void softmax_T(const float* __restrict__ lp,
                          unsigned short* __restrict__ dT) {
    const int k = blockIdx.x;                 // leaf index 0..511
    __shared__ float red[256];
    const float* row = lp + (long)k * NCLS;

    float m = -3.4e38f;
    for (int n = threadIdx.x; n < NCLS; n += 256) m = fmaxf(m, row[n]);
    red[threadIdx.x] = m; __syncthreads();
    for (int s = 128; s > 0; s >>= 1) {
        if (threadIdx.x < s) red[threadIdx.x] = fmaxf(red[threadIdx.x], red[threadIdx.x + s]);
        __syncthreads();
    }
    m = red[0]; __syncthreads();

    float sum = 0.0f;
    for (int n = threadIdx.x; n < NCLS; n += 256) sum += __expf(row[n] - m);
    red[threadIdx.x] = sum; __syncthreads();
    for (int s = 128; s > 0; s >>= 1) {
        if (threadIdx.x < s) red[threadIdx.x] += red[threadIdx.x + s];
        __syncthreads();
    }
    float inv = 1.0f / red[0];

    for (int n = threadIdx.x; n < NCLSP; n += 256) {
        float v = (n < NCLS) ? __expf(row[n] - m) * inv : 0.0f;
        dT[(long)n * LEAVES + k] = f2bf(v);
    }
}

// ---------------- tiled bf16 WMMA GEMM ----------------
// C[M x ncols] = A[M x K] * B^T  (B stored row-major N x K)
// block tile 128x128, 8 waves (wave32), each wave 32x64, K-step 32.
// Ping-pong LDS buffers; staging uses GLOBAL_LOAD_ASYNC_TO_LDS_B128 when
// available (ASYNCcnt), else global_load_b128 + ds_store_b128.
// EPI==1: C = sigmoid(acc + bias[n])   (ps for the tree)
// EPI==0: plain f32 store, masked to ncols
template <int EPI>
__global__ __launch_bounds__(256)
void gemm_bf16_wmma(const unsigned short* __restrict__ A, int lda,
                    const unsigned short* __restrict__ B, int ldb,
                    const float* __restrict__ bias, int nbias,
                    float* __restrict__ C, int ldc, int ncols, int K) {
    __shared__ __align__(16) unsigned short lA[2][128][32];
    __shared__ __align__(16) unsigned short lB[2][128][32];

    const int tid  = threadIdx.x;
    const int lane = tid & 31;
    const int wave = tid >> 5;                 // 0..7
    const int wm   = (wave & 3) * 32;          // wave M offset in tile
    const int wn   = (wave >> 2) * 64;         // wave N offset in tile
    const int m0   = blockIdx.x * 128;
    const int n0   = blockIdx.y * 128;

    const int ldr = tid >> 1;                  // 0..127 : row loaded by this thread
    const int ldk = (tid & 1) * 16;            // 0 or 16 : k-half

    // stage one 128x32 A tile and one 128x32 B tile into LDS buffer `buf`
    auto stage = [&](int buf, int k0) {
        const unsigned short* pa = A + (long)(m0 + ldr) * lda + (k0 + ldk);
        const unsigned short* pb = B + (long)(n0 + ldr) * ldb + (k0 + ldk);
#if HAVE_ASYNC_LDS
        __builtin_amdgcn_global_load_async_to_lds_b128(
            (GLOBAL_AS v4i*)pa,       (LOCAL_AS v4i*)&lA[buf][ldr][ldk],     0, 0);
        __builtin_amdgcn_global_load_async_to_lds_b128(
            (GLOBAL_AS v4i*)(pa + 8), (LOCAL_AS v4i*)&lA[buf][ldr][ldk + 8], 0, 0);
        __builtin_amdgcn_global_load_async_to_lds_b128(
            (GLOBAL_AS v4i*)pb,       (LOCAL_AS v4i*)&lB[buf][ldr][ldk],     0, 0);
        __builtin_amdgcn_global_load_async_to_lds_b128(
            (GLOBAL_AS v4i*)(pb + 8), (LOCAL_AS v4i*)&lB[buf][ldr][ldk + 8], 0, 0);
#else
        int4 a0 = *(const int4*)(pa);
        int4 a1 = *(const int4*)(pa + 8);
        int4 b0 = *(const int4*)(pb);
        int4 b1 = *(const int4*)(pb + 8);
        *(int4*)&lA[buf][ldr][ldk]     = a0;
        *(int4*)&lA[buf][ldr][ldk + 8] = a1;
        *(int4*)&lB[buf][ldr][ldk]     = b0;
        *(int4*)&lB[buf][ldr][ldk + 8] = b1;
#endif
    };

    v8f acc[2][4] = {};

    // prologue: stage first tile
    stage(0, 0);
#if HAVE_ASYNC_LDS
    WAIT_ASYNC();
#endif
    __syncthreads();

    int buf = 0;
    for (int k0 = 0; k0 < K; k0 += 32) {
        // issue async staging of the NEXT tile into the other buffer
        if (k0 + 32 < K) stage(buf ^ 1, k0 + 32);

        const int mr = lane & 15;
        const int kg = lane >> 4;

        FragBF fa[2], fb[4];
#pragma unroll
        for (int mi = 0; mi < 2; ++mi) {
            const int r = wm + mi * 16 + mr;
            fa[mi].q[0] = *(const int4*)&lA[buf][r][kg * 8];        // K = kg*8 .. +7
            fa[mi].q[1] = *(const int4*)&lA[buf][r][16 + kg * 8];   // K = 16+kg*8 .. +7
        }
#pragma unroll
        for (int ni = 0; ni < 4; ++ni) {
            const int r = wn + ni * 16 + mr;
            fb[ni].q[0] = *(const int4*)&lB[buf][r][kg * 8];        // K = kg*8 .. +7
            fb[ni].q[1] = *(const int4*)&lB[buf][r][16 + kg * 8];   // K = 16+kg*8 .. +7
        }

#pragma unroll
        for (int mi = 0; mi < 2; ++mi)
#pragma unroll
            for (int ni = 0; ni < 4; ++ni)
                acc[mi][ni] = __builtin_amdgcn_wmma_f32_16x16x32_bf16(
                    false, fa[mi].v, false, fb[ni].v,
                    (short)0, acc[mi][ni], false, false);

#if HAVE_ASYNC_LDS
        WAIT_ASYNC();                  // my async writes to buf^1 landed
#endif
        __syncthreads();               // everyone done reading buf + staging buf^1
        buf ^= 1;
    }

    // epilogue: C/D layout -> element (M = r + 8*(lane>>4), N = lane&15) in VGPR r
    const int en = lane & 15;
    const int eh = (lane >> 4) * 8;
#pragma unroll
    for (int mi = 0; mi < 2; ++mi) {
#pragma unroll
        for (int ni = 0; ni < 4; ++ni) {
#pragma unroll
            for (int r = 0; r < 8; ++r) {
                const int gm = m0 + wm + mi * 16 + eh + r;
                const int gn = n0 + wn + ni * 16 + en;
                float v = acc[mi][ni][r];
                if (EPI == 1) {
                    float bv = (gn < nbias) ? bias[gn] : 0.0f;
                    v += bv;
                    v = 1.0f / (1.0f + __expf(-v));            // sigmoid(BETA*x), BETA=1
                    C[(long)gm * ldc + gn] = v;
                } else {
                    if (gn < ncols) C[(long)gm * ldc + gn] = v;
                }
            }
        }
    }
}

// ---------------- tree path product ----------------
// one block per batch row; thread j computes leaf j's path product
__global__ __launch_bounds__(512)
void leaf_prob_kernel(const float* __restrict__ ps,
                      unsigned short* __restrict__ leafb) {
    const int row = blockIdx.x;
    __shared__ float sp[NODES];
    sp[threadIdx.x] = ps[(long)row * NODES + threadIdx.x];
    __syncthreads();

    const int j = threadIdx.x;                 // leaf 0..511
    float prod = 1.0f;
    int node = 0;
#pragma unroll
    for (int t = 0; t < LEVELS; ++t) {
        const int bit = (j >> (LEVELS - 1 - t)) & 1;
        const float p = sp[node];
        prod *= bit ? p : (1.0f - p);
        node = 2 * node + 1 + bit;
    }
    leafb[(long)row * LEAVES + j] = f2bf(prod);
}

// ---------------- host launcher ----------------
extern "C" void kernel_launch(void* const* d_in, const int* in_sizes, int n_in,
                              void* d_out, int out_size, void* d_ws, size_t ws_size,
                              hipStream_t stream) {
    const float* xs  = (const float*)d_in[0];   // 8192 x 4096
    const float* W   = (const float*)d_in[1];   // 511  x 4096
    const float* b   = (const float*)d_in[2];   // 511
    const float* lp  = (const float*)d_in[3];   // 512  x 1000
    float* out = (float*)d_out;                 // 8192 x 1000

    char* p = (char*)d_ws;
    unsigned short* xsb    = (unsigned short*)p; p += (size_t)BATCH * FEAT * 2;    // 67 MB
    unsigned short* Wb     = (unsigned short*)p; p += (size_t)NODES * FEAT * 2;    //  4 MB
    float*          ps     = (float*)p;          p += (size_t)BATCH * NODES * 4;   // 17 MB
    unsigned short* leafb  = (unsigned short*)p; p += (size_t)BATCH * LEAVES * 2;  //  8 MB
    unsigned short* distsT = (unsigned short*)p;                                   //  1 MB

    // 1) xs -> bf16
    {
        long n = (long)BATCH * FEAT;
        cvt_f32_bf16<<<(unsigned)((n + 255) / 256), 256, 0, stream>>>(xs, xsb, n);
    }
    // 2) W -> bf16 (padded to 512 rows)
    cvt_W<<<(NODES * FEAT) / 256, 256, 0, stream>>>(W, Wb);
    // 3) softmax(leaf_params), transposed + padded
    softmax_T<<<LEAVES, 256, 0, stream>>>(lp, distsT);
    // 4) GEMM1 + bias + sigmoid -> ps (8192 x 512)
    gemm_bf16_wmma<1><<<dim3(BATCH / 128, NODES / 128), 256, 0, stream>>>(
        xsb, FEAT, Wb, FEAT, b, NBIAS, ps, NODES, NODES, FEAT);
    // 5) tree path products -> leaf probabilities (bf16)
    leaf_prob_kernel<<<BATCH, 512, 0, stream>>>(ps, leafb);
    // 6) GEMM2 -> out (8192 x 1000)
    gemm_bf16_wmma<0><<<dim3(BATCH / 128, NCLSP / 128), 256, 0, stream>>>(
        leafb, LEAVES, distsT, LEAVES, nullptr, 0, out, NCLS, NCLS, LEAVES);
}
